// MemEffAttention_54185307407078
// MI455X (gfx1250) — matrix operand order repaired and checked by reading, not compiled, not run on hardware
//
#include <hip/hip_runtime.h>
#include <hip/hip_bf16.h>
#include <math.h>

// ---------------- problem constants ----------------
#define N_TOK 2309
#define CDIM  1024
#define HEADS 16
#define DHEAD 64
#define REGQ  5
#define NPAD  2336          // key count padded to multiple of 32 (>= ceil(N/32)*32)
#define QTILES   ((N_TOK + 15) / 16)   // 145 (attention query tiles)
#define NT32     ((N_TOK + 31) / 32)   // 73  (GEMM 32-row tiles)

typedef __attribute__((ext_vector_type(16))) __bf16 v16bf;
typedef __attribute__((ext_vector_type(8)))  float  v8f;

union ABu { uint4 u[2]; v16bf v; };

__device__ __forceinline__ v8f wmma_bf16(v16bf a, v16bf b, v8f c) {
    // D = A(16x32 bf16) * B(32x16 bf16) + C(16x16 f32)
    return __builtin_amdgcn_wmma_f32_16x16x32_bf16(false, a, false, b, (short)0, c, false, false);
}

// A fragment (16x32, MxK) from row-major bf16 matrix, leading dim `ld`.
// lane<16 : row=row0+lane,    K = kbase+[0..7], kbase+[16..23]
// lane>=16: row=row0+lane-16, K = kbase+[8..15], kbase+[24..31]
__device__ __forceinline__ v16bf load_frag_A(const __bf16* base, int row0, int ld,
                                             int kbase, int rowMax) {
    int lane = threadIdx.x & 31;
    int r = row0 + (lane & 15);
    if (r > rowMax) r = rowMax;
    int lo = kbase + ((lane & 16) ? 8 : 0);
    const __bf16* p = base + (size_t)r * ld + lo;
    ABu t;
    t.u[0] = *(const uint4*)(p);
    t.u[1] = *(const uint4*)(p + 16);
    return t.v;
}

// B fragment (32x16, KxN) from memory where each column's contraction run is
// contiguous: elem(col, k) at base[col*ld + k].
// lane<16 : col=col0+lane,    K = kbase+[0..7], kbase+[16..23]
// lane>=16: col=col0+lane-16, K = kbase+[8..15], kbase+[24..31]
__device__ __forceinline__ v16bf load_frag_B(const __bf16* base, int col0, int ld,
                                             int kbase, int colMax) {
    int lane = threadIdx.x & 31;
    int c = col0 + (lane & 15);
    if (c > colMax) c = colMax;
    int kb = kbase + ((lane & 16) ? 8 : 0);
    const __bf16* p = base + (size_t)c * ld + kb;
    ABu t;
    t.u[0] = *(const uint4*)(p);
    t.u[1] = *(const uint4*)(p + 16);
    return t.v;
}

// ---------------- software-pipelined 32x64 GEMM core (K = CDIM) ----------------
// Ping-pong fragment double buffering: next-chunk loads issue while current
// WMMAs execute, so loadcnt waits become partial instead of full barriers.
__device__ __forceinline__ void gemm32x64(const __bf16* __restrict__ Ab, int row0, int rowMax,
                                          const __bf16* __restrict__ Bb, int col0, int colMax,
                                          v8f acc[2][4]) {
    v16bf a0[2], a1[2], b0[4], b1[4];
#pragma unroll
    for (int i = 0; i < 2; i++) a0[i] = load_frag_A(Ab, row0 + i * 16, CDIM, 0, rowMax);
#pragma unroll
    for (int t = 0; t < 4; t++) b0[t] = load_frag_B(Bb, col0 + t * 16, CDIM, 0, colMax);
    for (int kc = 0; kc < CDIM; kc += 64) {
        // stage kc+32 while computing kc
        __builtin_prefetch(Ab + (size_t)(row0 + (threadIdx.x & 15)) * CDIM + kc + 128, 0, 3);
#pragma unroll
        for (int i = 0; i < 2; i++) a1[i] = load_frag_A(Ab, row0 + i * 16, CDIM, kc + 32, rowMax);
#pragma unroll
        for (int t = 0; t < 4; t++) b1[t] = load_frag_B(Bb, col0 + t * 16, CDIM, kc + 32, colMax);
#pragma unroll
        for (int i = 0; i < 2; i++)
#pragma unroll
            for (int t = 0; t < 4; t++) acc[i][t] = wmma_bf16(a0[i], b0[t], acc[i][t]);
        if (kc + 64 < CDIM) {   // stage kc+64 while computing kc+32
#pragma unroll
            for (int i = 0; i < 2; i++) a0[i] = load_frag_A(Ab, row0 + i * 16, CDIM, kc + 64, rowMax);
#pragma unroll
            for (int t = 0; t < 4; t++) b0[t] = load_frag_B(Bb, col0 + t * 16, CDIM, kc + 64, colMax);
        }
#pragma unroll
        for (int i = 0; i < 2; i++)
#pragma unroll
            for (int t = 0; t < 4; t++) acc[i][t] = wmma_bf16(a1[i], b1[t], acc[i][t]);
    }
}

// ---------------- precision conversion ----------------
__global__ void cvt_f32_bf16(const float* __restrict__ in, __bf16* __restrict__ out, int n) {
    int i = blockIdx.x * blockDim.x + threadIdx.x;
    int stride = gridDim.x * blockDim.x;
    for (; i < n; i += stride) out[i] = (__bf16)in[i];
}

// out[c][r] = in[r][c]  (in: rows x cols row-major, out: cols x rows)
__global__ void cvtT_f32_bf16(const float* __restrict__ in, __bf16* __restrict__ out,
                              int rows, int cols) {
    int n = rows * cols;
    int i = blockIdx.x * blockDim.x + threadIdx.x;
    int stride = gridDim.x * blockDim.x;
    for (; i < n; i += stride) {
        int r = i / cols, c = i % cols;
        out[(size_t)c * rows + r] = (__bf16)in[i];
    }
}

// ---------------- QKV GEMM: x(N,C) @ WqkvT -> q scaled / k bf16 (H,N,D), v bf16 (H,D,NPAD)
__global__ __launch_bounds__(32) void qkv_gemm(const __bf16* __restrict__ xb,
                                               const __bf16* __restrict__ wT,   // (3C, C)
                                               __bf16* __restrict__ qb,
                                               __bf16* __restrict__ kb,
                                               __bf16* __restrict__ vt) {
    int nb = blockIdx.x;          // 32-row tile
    int cb = blockIdx.y;          // 64-col tile (48 of them)
    int lane = threadIdx.x & 31;
    v8f acc[2][4] = {};
    gemm32x64(xb, nb * 32, N_TOK - 1, wT, cb * 64, 3 * CDIM - 1, acc);

    int part = cb >> 4;            // 0=q 1=k 2=v
    int h = cb & 15;
    float sc = (part == 0) ? 0.125f : 1.0f;   // D^-0.5
#pragma unroll
    for (int i = 0; i < 2; i++) {
#pragma unroll
        for (int t = 0; t < 4; t++) {
            int d = t * 16 + (lane & 15);
#pragma unroll
            for (int r = 0; r < 8; r++) {
                int row = nb * 32 + i * 16 + ((lane & 16) ? r + 8 : r);
                if (row < N_TOK) {
                    __bf16 val = (__bf16)(acc[i][t][r] * sc);
                    if (part == 0)      qb[((size_t)h * N_TOK + row) * DHEAD + d] = val;
                    else if (part == 1) kb[((size_t)h * N_TOK + row) * DHEAD + d] = val;
                    else                vt[((size_t)h * DHEAD + d) * NPAD + row]  = val;
                }
            }
        }
    }
}

// ---------------- cosine similarity vs g probe: one block per pure token ----------------
__global__ __launch_bounds__(512) void sim_kernel(const __bf16* __restrict__ qb,  // (H,N,D)
                                                  const float* __restrict__ g,    // g_info flat
                                                  float* __restrict__ sim) {
    int n = blockIdx.x + REGQ;     // absolute token
    int w = threadIdx.x >> 5;      // head
    int l = threadIdx.x & 31;
    __shared__ float hs[HEADS];
    float dot = 0.f, qq = 0.f, gg = 0.f;
    for (int d = l; d < DHEAD; d += 32) {
        float qv = (float)qb[((size_t)w * N_TOK + n) * DHEAD + d];
        float gv = g[w * DHEAD + d];             // g_info[0,0,:] reshaped (H,D)
        dot += qv * gv; qq += qv * qv; gg += gv * gv;
    }
#pragma unroll
    for (int off = 16; off > 0; off >>= 1) {
        dot += __shfl_xor(dot, off);
        qq  += __shfl_xor(qq,  off);
        gg  += __shfl_xor(gg,  off);
    }
    if (l == 0) hs[w] = dot * rsqrtf(qq * gg + 1e-30f);
    __syncthreads();
    if (threadIdx.x == 0) {
        float s = 0.f;
#pragma unroll
        for (int i = 0; i < HEADS; i++) s += hs[i];
        sim[blockIdx.x] = s * (1.0f / HEADS);
    }
}

// ---------------- min/max normalize + threshold ----------------
__global__ __launch_bounds__(1024) void pos_kernel(const float* __restrict__ sim, int M,
                                                   int* __restrict__ pos) {
    __shared__ float sA[32], sB[32];
    int tid = threadIdx.x;
    float mn = INFINITY, mx = -INFINITY;
    for (int i = tid; i < M; i += 1024) {
        float v = sim[i];
        mn = fminf(mn, v); mx = fmaxf(mx, v);
    }
#pragma unroll
    for (int off = 16; off > 0; off >>= 1) {
        mn = fminf(mn, __shfl_xor(mn, off));
        mx = fmaxf(mx, __shfl_xor(mx, off));
    }
    if ((tid & 31) == 0) { sA[tid >> 5] = mn; sB[tid >> 5] = mx; }
    __syncthreads();
    if (tid < 32) {
        mn = sA[tid]; mx = sB[tid];
#pragma unroll
        for (int off = 16; off > 0; off >>= 1) {
            mn = fminf(mn, __shfl_xor(mn, off));
            mx = fmaxf(mx, __shfl_xor(mx, off));
        }
        if (tid == 0) { sA[0] = mn; sB[0] = mx; }
    }
    __syncthreads();
    float gmn = sA[0];
    float inv = 1.0f / (sB[0] - gmn + 1e-20f);
    for (int i = tid; i < M; i += 1024)
        pos[i] = ((sim[i] - gmn) * inv > 0.9f) ? 1 : 0;
}

// ---------------- dual-mask flash pass over one 32-key chunk ----------------
__device__ __forceinline__ void flash_pass(const v8f& S0, const v8f& S1, bool inc0, bool inc1,
                                           float m[8], float l[8], v8f o[4],
                                           const v16bf bV[4], __bf16* pl, int lane) {
    float s0[8], s1[8], cm[8];
#pragma unroll
    for (int r = 0; r < 8; r++) {
        s0[r] = inc0 ? S0[r] : -INFINITY;
        s1[r] = inc1 ? S1[r] : -INFINITY;
        cm[r] = fmaxf(s0[r], s1[r]);
    }
#pragma unroll
    for (int off = 1; off < 16; off <<= 1)
#pragma unroll
        for (int r = 0; r < 8; r++) cm[r] = fmaxf(cm[r], __shfl_xor(cm[r], off));
    float f[8], rs[8];
#pragma unroll
    for (int r = 0; r < 8; r++) {
        float mo = m[r];
        float nm = fmaxf(mo, cm[r]);
        f[r] = (nm == mo) ? 1.0f : ((mo == -INFINITY) ? 0.0f : __expf(mo - nm));
        m[r] = nm;
        float p0 = (s0[r] == -INFINITY) ? 0.0f : __expf(s0[r] - nm);
        float p1 = (s1[r] == -INFINITY) ? 0.0f : __expf(s1[r] - nm);
        int rowabs = (lane & 16) ? r + 8 : r;
        pl[rowabs * 32 + (lane & 15)]      = (__bf16)p0;
        pl[rowabs * 32 + 16 + (lane & 15)] = (__bf16)p1;
        rs[r] = p0 + p1;
    }
#pragma unroll
    for (int off = 1; off < 16; off <<= 1)
#pragma unroll
        for (int r = 0; r < 8; r++) rs[r] += __shfl_xor(rs[r], off);
#pragma unroll
    for (int r = 0; r < 8; r++) l[r] = l[r] * f[r] + rs[r];
    __syncthreads();                         // LDS stores visible
    v16bf aP = load_frag_A(pl, 0, 32, 0, 15);
    __syncthreads();                         // reads done before next overwrite
#pragma unroll
    for (int t = 0; t < 4; t++) {
#pragma unroll
        for (int r = 0; r < 8; r++) o[t][r] *= f[r];
        o[t] = wmma_bf16(aP, bV[t], o[t]);
    }
}

// ---------------- attention: one wave per (head, 16-query tile) ----------------
__global__ __launch_bounds__(32) void attn_kernel(const __bf16* __restrict__ qb,  // (H,N,D) scaled
                                                  const __bf16* __restrict__ kb,  // (H,N,D)
                                                  const __bf16* __restrict__ vt,  // (H,D,NPAD)
                                                  const int* __restrict__ pos,
                                                  __bf16* __restrict__ ctx) {     // (N,C)
    int h  = blockIdx.y;
    int q0 = blockIdx.x * 16;
    int lane = threadIdx.x & 31;
    const __bf16* Qh = qb + (size_t)h * N_TOK * DHEAD;
    const __bf16* Kh = kb + (size_t)h * N_TOK * DHEAD;
    const __bf16* Vh = vt + (size_t)h * DHEAD * NPAD;
    __shared__ __align__(16) __bf16 pl[16 * 32];

    v16bf aQ0 = load_frag_A(Qh, q0, DHEAD, 0,  N_TOK - 1);
    v16bf aQ1 = load_frag_A(Qh, q0, DHEAD, 32, N_TOK - 1);

    v8f oP[4] = {}, oN[4] = {};
    float mP[8], lP[8], mN[8], lN[8];
#pragma unroll
    for (int r = 0; r < 8; r++) { mP[r] = -INFINITY; mN[r] = -INFINITY; lP[r] = 0.f; lN[r] = 0.f; }

    const int nch = (N_TOK + 31) / 32;
    for (int c = 0; c < nch; c++) {
        int key0 = c * 32;
        // --- logits S = Q * K^T for 32 keys (two 16-key subtiles)
        v8f S0 = {}, S1 = {};
        {
            v16bf b00 = load_frag_B(Kh, key0,      DHEAD, 0,  N_TOK - 1);
            v16bf b01 = load_frag_B(Kh, key0,      DHEAD, 32, N_TOK - 1);
            v16bf b10 = load_frag_B(Kh, key0 + 16, DHEAD, 0,  N_TOK - 1);
            v16bf b11 = load_frag_B(Kh, key0 + 16, DHEAD, 32, N_TOK - 1);
            S0 = wmma_bf16(aQ0, b00, S0); S0 = wmma_bf16(aQ1, b01, S0);
            S1 = wmma_bf16(aQ0, b10, S1); S1 = wmma_bf16(aQ1, b11, S1);
        }
        // --- V fragments for this chunk (shared by both masked passes)
        v16bf bV[4];
#pragma unroll
        for (int t = 0; t < 4; t++) bV[t] = load_frag_B(Vh, t * 16, NPAD, key0, DHEAD - 1);

        // --- per-lane key masks (lane owns one key column per subtile)
        int j0 = key0 + (lane & 15);
        int j1 = j0 + 16;
        bool ip0 = (j0 < REGQ) || (j0 < N_TOK && pos[j0 - REGQ] != 0);
        bool ip1 = (j1 < REGQ) || (j1 < N_TOK && pos[j1 - REGQ] != 0);
        bool in0 = (j0 < REGQ) || (j0 < N_TOK && pos[j0 - REGQ] == 0);
        bool in1 = (j1 < REGQ) || (j1 < N_TOK && pos[j1 - REGQ] == 0);

        flash_pass(S0, S1, ip0, ip1, mP, lP, oP, bV, pl, lane);
        flash_pass(S0, S1, in0, in1, mN, lN, oN, bV, pl, lane);
    }

    // --- finalize: normalize, combine pos/neg per query, store bf16 context
#pragma unroll
    for (int r = 0; r < 8; r++) {
        int rowabs = q0 + ((lane & 16) ? r + 8 : r);
        if (rowabs >= N_TOK) continue;
        float ivP = (lP[r] > 0.f) ? 1.0f / lP[r] : 0.0f;
        float ivN = (lN[r] > 0.f) ? 1.0f / lN[r] : 0.0f;
        bool qp = (rowabs >= REGQ) ? (pos[rowabs - REGQ] != 0) : false;
#pragma unroll
        for (int t = 0; t < 4; t++) {
            float vp = oP[t][r] * ivP;
            float vn = oN[t][r] * ivN;
            float ov = (rowabs < REGQ) ? 0.5f * (vp + vn) : (qp ? vp : vn);
            ctx[(size_t)rowabs * CDIM + h * DHEAD + t * 16 + (lane & 15)] = (__bf16)ov;
        }
    }
}

// ---------------- output projection: ctx(N,C) @ WprojT + bias -> fp32 out ----------------
__global__ __launch_bounds__(32) void proj_gemm(const __bf16* __restrict__ ctx,
                                                const __bf16* __restrict__ wT,   // (C, C) transposed
                                                const float* __restrict__ bias,
                                                float* __restrict__ out) {
    int nb = blockIdx.x, cb = blockIdx.y;
    int lane = threadIdx.x & 31;
    v8f acc[2][4] = {};
    gemm32x64(ctx, nb * 32, N_TOK - 1, wT, cb * 64, CDIM - 1, acc);

#pragma unroll
    for (int i = 0; i < 2; i++) {
#pragma unroll
        for (int t = 0; t < 4; t++) {
            int col = cb * 64 + t * 16 + (lane & 15);
            float bv = bias[col];
#pragma unroll
            for (int r = 0; r < 8; r++) {
                int row = nb * 32 + i * 16 + ((lane & 16) ? r + 8 : r);
                if (row < N_TOK) out[(size_t)row * CDIM + col] = acc[i][t][r] + bv;
            }
        }
    }
}

// ---------------- host entry ----------------
extern "C" void kernel_launch(void* const* d_in, const int* in_sizes, int n_in,
                              void* d_out, int out_size, void* d_ws, size_t ws_size,
                              hipStream_t stream) {
    (void)in_sizes; (void)n_in; (void)out_size; (void)ws_size;
    const float* x     = (const float*)d_in[0];
    const float* g     = (const float*)d_in[1];
    const float* Wqkv  = (const float*)d_in[2];
    const float* Wproj = (const float*)d_in[3];
    const float* bproj = (const float*)d_in[4];
    float* out = (float*)d_out;

    char* ws = (char*)d_ws;
    size_t off = 0;
    auto alloc = [&](size_t bytes) -> char* {
        char* p = ws + off;
        off = (off + bytes + 255) & ~(size_t)255;
        return p;
    };
    __bf16* xb  = (__bf16*)alloc((size_t)N_TOK * CDIM * 2);
    __bf16* wqT = (__bf16*)alloc((size_t)3 * CDIM * CDIM * 2);
    __bf16* wpT = (__bf16*)alloc((size_t)CDIM * CDIM * 2);
    __bf16* qb  = (__bf16*)alloc((size_t)HEADS * N_TOK * DHEAD * 2);
    __bf16* kb  = (__bf16*)alloc((size_t)HEADS * N_TOK * DHEAD * 2);
    __bf16* vt  = (__bf16*)alloc((size_t)HEADS * DHEAD * NPAD * 2);
    __bf16* ctx = (__bf16*)alloc((size_t)N_TOK * CDIM * 2);
    float*  sim = (float*)alloc((size_t)(N_TOK - REGQ) * 4);
    int*    pos = (int*)alloc((size_t)(N_TOK - REGQ) * 4);

    cvt_f32_bf16<<<2048, 256, 0, stream>>>(x, xb, N_TOK * CDIM);
    cvtT_f32_bf16<<<2048, 256, 0, stream>>>(Wqkv, wqT, CDIM, 3 * CDIM);
    cvtT_f32_bf16<<<2048, 256, 0, stream>>>(Wproj, wpT, CDIM, CDIM);

    qkv_gemm<<<dim3(NT32, 48), 32, 0, stream>>>(xb, wqT, qb, kb, vt);
    sim_kernel<<<N_TOK - REGQ, 512, 0, stream>>>(qb, g, sim);
    pos_kernel<<<1, 1024, 0, stream>>>(sim, N_TOK - REGQ, pos);
    attn_kernel<<<dim3(QTILES, HEADS), 32, 0, stream>>>(qb, kb, vt, pos, ctx);
    proj_gemm<<<dim3(NT32, HEADS), 32, 0, stream>>>(ctx, wpT, bproj, out);
}
